// Attention_31808527794412
// MI455X (gfx1250) — matrix operand order repaired and checked by reading, not compiled, not run on hardware
//
#include <hip/hip_runtime.h>

typedef __bf16 bf16_t;
typedef __attribute__((ext_vector_type(16))) __bf16 v16bf;
typedef __attribute__((ext_vector_type(2)))  __bf16 v2bf;
typedef __attribute__((ext_vector_type(8)))  float  v8f;
typedef __attribute__((ext_vector_type(2)))  float  v2f;

#define DIM   1024
#define HEADS 16
#define HD    64
#define BATCH 4
#define SEQ   2048
#define MTOT  (BATCH * SEQ)   // 8192

// softmax scale (HD^-0.5) pre-multiplied by log2(e): flash works in exp2 domain
#define QSCALE_LOG2E 0.18033688011112042f

__device__ __forceinline__ v8f wmma_bf16(v16bf a, v16bf b, v8f c) {
  return __builtin_amdgcn_wmma_f32_16x16x32_bf16(
      /*neg_a=*/false, a, /*neg_b=*/false, b,
      /*c_mod=*/(short)0, c, /*reuse_a=*/false, /*reuse_b=*/false);
}

__device__ __forceinline__ float fast_exp2(float x) {
#if __has_builtin(__builtin_amdgcn_exp2f)
  return __builtin_amdgcn_exp2f(x);
#else
  return exp2f(x);
#endif
}

// pack two f32 -> packed bf16 pair (backend picks v_cvt_pk_bf16_f32)
__device__ __forceinline__ unsigned pack2(float a, float b) {
  v2f s; s.x = a; s.y = b;
  v2bf h = __builtin_convertvector(s, v2bf);
  union { v2bf h; unsigned u; } t;
  t.h = h;
  return t.u;
}

// A-fragment (16x32 bf16), row-major source: element (m,k) at base[m*ld + k].
// lanes 0-15 row M=lane, K=0..15; lanes 16-31 K=16..31 (contiguous, matches B).
__device__ __forceinline__ v16bf load_a_rowmajor(const bf16_t* base, int ld) {
  const int lane = threadIdx.x & 31;
  const int row  = lane & 15;
  const int kb   = (lane >> 4) << 4;           // 0 or 16
  const bf16_t* p = base + row * ld + kb;
  v16bf f;
#pragma unroll
  for (int i = 0; i < 8; ++i) f[i] = p[i];
#pragma unroll
  for (int i = 0; i < 8; ++i) f[8 + i] = p[8 + i];
  return f;
}

// A-fragment from column-major source: element (m,k) at base[k*ld + m].
__device__ __forceinline__ v16bf load_a_colmajor(const bf16_t* base, int ld) {
  const int lane = threadIdx.x & 31;
  const int row  = lane & 15;
  const int kb   = (lane >> 4) << 4;
  v16bf f;
#pragma unroll
  for (int i = 0; i < 8; ++i) f[i] = base[(kb + i) * ld + row];
#pragma unroll
  for (int i = 0; i < 8; ++i) f[8 + i] = base[(8 + kb + i) * ld + row];
  return f;
}

// B-fragment (32x16 bf16) from n-major source: element (k,n) at base[n*ld + k].
// contiguous k order (same as A loaders): lanes 0-15 hold K=0..15 (col=lane), lanes 16-31 hold K=16..31.
__device__ __forceinline__ v16bf load_b_nmajor(const bf16_t* base, int ld) {
  const int lane = threadIdx.x & 31;
  const int col  = lane & 15;
  const int koff = (lane >> 4) << 4;           // 0 or 16
  const bf16_t* p = base + col * ld + koff;
  v16bf f;
#pragma unroll
  for (int i = 0; i < 16; ++i) f[i] = p[i];
  return f;
}

// ---------------------------------------------------------------------------
// Kernel 1: qkv = x @ qkv_w + qkv_b ; scatter to Q(*scale)/K/V [B,H,SEQ,HD] bf16
// ---------------------------------------------------------------------------
__global__ __launch_bounds__(256) void qkv_gemm_kernel(
    const float* __restrict__ X, const float* __restrict__ W,
    const float* __restrict__ bias,
    bf16_t* __restrict__ Qo, bf16_t* __restrict__ Ko, bf16_t* __restrict__ Vo) {
  __shared__ alignas(16) bf16_t As[128][40];   // [m][k], pad 8
  __shared__ alignas(16) bf16_t Bs[128][40];   // [n][k] (transposed), pad 8
  const int mbase = blockIdx.x * 128;
  const int nbase = blockIdx.y * 128;
  const int tid   = threadIdx.x;
  const int wave  = tid >> 5;
  const int lane  = tid & 31;
  const int wm    = wave >> 2;                  // 0..1  (64 rows each)
  const int wn    = wave & 3;                   // 0..3  (32 cols each)

  v8f acc[4][2];
#pragma unroll
  for (int i = 0; i < 4; ++i)
#pragma unroll
    for (int j = 0; j < 2; ++j) acc[i][j] = {};

  for (int k0 = 0; k0 < DIM; k0 += 32) {
    // A tile: 128x32 fp32 -> bf16, packed b64 LDS stores
#pragma unroll
    for (int p = 0; p < 4; ++p) {
      const int r = (tid >> 3) + p * 32;
      const int c = (tid & 7) << 2;
      const float4 va = *(const float4*)(X + (size_t)(mbase + r) * DIM + k0 + c);
      uint2 w;
      w.x = pack2(va.x, va.y);
      w.y = pack2(va.z, va.w);
      *(uint2*)&As[r][c] = w;
    }
    // B tile: 32x128 fp32 -> bf16 transposed [n][k]; pack (k,k+1) pairs -> b32 stores
#pragma unroll
    for (int p = 0; p < 2; ++p) {
      const int k = ((tid >> 5) + p * 8) << 1;  // even k row, 0..30
      const int n4 = (tid & 31) << 2;
      const float4 w0 = *(const float4*)(W + (size_t)(k0 + k) * (3 * DIM) + nbase + n4);
      const float4 w1 = *(const float4*)(W + (size_t)(k0 + k + 1) * (3 * DIM) + nbase + n4);
      *(unsigned*)&Bs[n4 + 0][k] = pack2(w0.x, w1.x);
      *(unsigned*)&Bs[n4 + 1][k] = pack2(w0.y, w1.y);
      *(unsigned*)&Bs[n4 + 2][k] = pack2(w0.z, w1.z);
      *(unsigned*)&Bs[n4 + 3][k] = pack2(w0.w, w1.w);
    }
    __syncthreads();
    v16bf af[4], bfr[2];
#pragma unroll
    for (int i = 0; i < 4; ++i) af[i] = load_a_rowmajor(&As[wm * 64 + 16 * i][0], 40);
#pragma unroll
    for (int j = 0; j < 2; ++j) bfr[j] = load_b_nmajor(&Bs[wn * 32 + 16 * j][0], 40);
#pragma unroll
    for (int i = 0; i < 4; ++i)
#pragma unroll
      for (int j = 0; j < 2; ++j) acc[i][j] = wmma_bf16(af[i], bfr[j], acc[i][j]);
    __syncthreads();
  }

  // Epilogue: bias, scale q (scale * log2e folded in), scatter to [B,H,SEQ,HD]
  const int colq  = lane & 15;
  const int rhalf = (lane >> 4) << 3;
  bf16_t* dsts[3] = {Qo, Ko, Vo};
#pragma unroll
  for (int i = 0; i < 4; ++i) {
#pragma unroll
    for (int j = 0; j < 2; ++j) {
      const int n = nbase + wn * 32 + 16 * j + colq;  // 0..3071
      const int which = n >> 10;
      const int rem = n & 1023;
      const int h = rem >> 6, d = rem & 63;
      const float bv = bias[n];
      const float sc = (which == 0) ? QSCALE_LOG2E : 1.0f;
      bf16_t* dst = dsts[which];
#pragma unroll
      for (int r8 = 0; r8 < 8; ++r8) {
        const int m = mbase + wm * 64 + 16 * i + rhalf + r8;
        const int bb = m >> 11, nr = m & 2047;
        const float v = (acc[i][j][r8] + bv) * sc;
        dst[(((size_t)bb * HEADS + h) * SEQ + nr) * HD + d] = (bf16_t)v;
      }
    }
  }
}

// ---------------------------------------------------------------------------
// Kernel 2: flash attention. Grid (B*H, SEQ/128), 8 waves, 16 queries/wave.
// K/V tiles streamed via GLOBAL_LOAD_ASYNC_TO_LDS_B128 (ASYNCcnt), double
// buffered so the DMA overlaps WMMA + online softmax on the current tile.
// S^T = K.Q^T (keys in C-vgpr dim, queries in lanes) -> lane-local softmax
// in exp2 domain. O^T += V^T.P^T, 4 f32 C-tiles. Output [B,SEQ,DIM] bf16.
// ---------------------------------------------------------------------------
__device__ __forceinline__ void async_kv_tile(const bf16_t* Kb, const bf16_t* Vb,
                                              int k0, int r, int c,
                                              unsigned ldsK, unsigned ldsV) {
  const unsigned long long gK = (unsigned long long)(Kb + (size_t)(k0 + r) * HD + c);
  const unsigned long long gV = (unsigned long long)(Vb + (size_t)(k0 + r) * HD + c);
  asm volatile("global_load_async_to_lds_b128 %0, %1, off"
               :: "v"(ldsK), "v"(gK) : "memory");
  asm volatile("global_load_async_to_lds_b128 %0, %1, off"
               :: "v"(ldsV), "v"(gV) : "memory");
}

__global__ __launch_bounds__(256) void flash_attn_kernel(
    const bf16_t* __restrict__ Q, const bf16_t* __restrict__ K,
    const bf16_t* __restrict__ V, bf16_t* __restrict__ O) {
  __shared__ alignas(16) bf16_t Ks[2][32][72];   // double-buffered 32 keys x 64 d (+8 pad)
  __shared__ alignas(16) bf16_t Vs[2][32][72];
  const int bh   = blockIdx.x;                 // b*HEADS + h
  const int b    = bh >> 4;
  const int h    = bh & 15;
  const int wave = threadIdx.x >> 5;
  const int lane = threadIdx.x & 31;
  const int qrow = blockIdx.y * 128 + wave * 16;
  const size_t bhOff = (size_t)bh * SEQ * HD;
  const bf16_t* Qb = Q + bhOff;
  const bf16_t* Kb = K + bhOff;
  const bf16_t* Vb = V + bhOff;

  // per-thread slot in the cooperative 32x64 tile copy
  const int tr = threadIdx.x >> 3;             // 0..31 keys
  const int tc = (threadIdx.x & 7) << 3;       // 0..56, 8 bf16
  const unsigned ldsK0 = (unsigned)(unsigned long long)&Ks[0][tr][tc];
  const unsigned ldsK1 = (unsigned)(unsigned long long)&Ks[1][tr][tc];
  const unsigned ldsV0 = (unsigned)(unsigned long long)&Vs[0][tr][tc];
  const unsigned ldsV1 = (unsigned)(unsigned long long)&Vs[1][tr][tc];

  // Q^T B-fragments (constant across key loop): d chunks 0..31, 32..63
  const v16bf qf0 = load_b_nmajor(Qb + (size_t)qrow * HD + 0, HD);
  const v16bf qf1 = load_b_nmajor(Qb + (size_t)qrow * HD + 32, HD);

  v8f acc[4];                                   // O^T tiles: d 0-15,16-31,32-47,48-63
#pragma unroll
  for (int t = 0; t < 4; ++t) acc[t] = {};
  float m_run = -1e30f, l_run = 0.0f;

  // prime buffer 0
  async_kv_tile(Kb, Vb, 0, tr, tc, ldsK0, ldsV0);
  asm volatile("s_wait_asynccnt 0x0" ::: "memory");
  __syncthreads();

  int cur = 0;
  for (int k0 = 0; k0 < SEQ; k0 += 32) {
    // prefetch next tile into the other buffer (overlaps with compute below)
    if (k0 + 32 < SEQ)
      async_kv_tile(Kb, Vb, k0 + 32, tr, tc,
                    cur ? ldsK0 : ldsK1, cur ? ldsV0 : ldsV1);

    const bf16_t* Kt = &Ks[cur][0][0];
    const bf16_t* Vt = &Vs[cur][0][0];

    // S^T tiles: st0 = keys 0-15, st1 = keys 16-31 (relative)
    v8f st0 = {}, st1 = {};
    st0 = wmma_bf16(load_a_rowmajor(Kt + 0 * 72 + 0, 72), qf0, st0);
    st0 = wmma_bf16(load_a_rowmajor(Kt + 0 * 72 + 32, 72), qf1, st0);
    st1 = wmma_bf16(load_a_rowmajor(Kt + 16 * 72 + 0, 72), qf0, st1);
    st1 = wmma_bf16(load_a_rowmajor(Kt + 16 * 72 + 32, 72), qf1, st1);

    // online softmax, exp2 domain (per query column; combine partner half-wave)
    float smax = st0[0];
#pragma unroll
    for (int i = 1; i < 8; ++i) smax = fmaxf(smax, st0[i]);
#pragma unroll
    for (int i = 0; i < 8; ++i) smax = fmaxf(smax, st1[i]);
    smax = fmaxf(smax, __shfl_xor(smax, 16, 32));
    const float m_new = fmaxf(m_run, smax);
    const float corr  = fast_exp2(m_run - m_new);
    float p0[8], p1[8], psum = 0.0f;
#pragma unroll
    for (int i = 0; i < 8; ++i) { p0[i] = fast_exp2(st0[i] - m_new); psum += p0[i]; }
#pragma unroll
    for (int i = 0; i < 8; ++i) { p1[i] = fast_exp2(st1[i] - m_new); psum += p1[i]; }
    psum += __shfl_xor(psum, 16, 32);
    l_run = l_run * corr + psum;
    m_run = m_new;
#pragma unroll
    for (int t = 0; t < 4; ++t)
#pragma unroll
      for (int i = 0; i < 8; ++i) acc[t][i] *= corr;

    // Build P^T B-fragment. Pack own p-values to bf16 pairs first, then a
    // single 32-bit shfl_xor(16) per dword swaps halves with the partner lane:
    // lane<16 needs keys 0..15 = {own st0, partner st0};
    // lane>=16 needs keys 16..31 = {partner st1, own st1}.
    unsigned u0[4], u1[4];
#pragma unroll
    for (int j = 0; j < 4; ++j) {
      u0[j] = pack2(p0[2 * j], p0[2 * j + 1]);
      u1[j] = pack2(p1[2 * j], p1[2 * j + 1]);
    }
    union { v16bf f; unsigned u[8]; } pf;
#pragma unroll
    for (int j = 0; j < 4; ++j) {
      const unsigned sel = (lane < 16) ? u1[j] : u0[j];
      const unsigned got = __shfl_xor(sel, 16, 32);
      pf.u[j]     = (lane < 16) ? u0[j] : got;   // k = koff + 2j, 2j+1
      pf.u[4 + j] = (lane < 16) ? got : u1[j];   // k = koff + 8 + 2j, 2j+1
    }

    // O^T += V^T . P^T  (4 d-tiles)
#pragma unroll
    for (int t = 0; t < 4; ++t)
      acc[t] = wmma_bf16(load_a_colmajor(Vt + 16 * t, 72), pf.f, acc[t]);

    // next tile's async copy must land, and all waves must be done reading cur
    asm volatile("s_wait_asynccnt 0x0" ::: "memory");
    __syncthreads();
    cur ^= 1;
  }

  const float inv = 1.0f / l_run;
  const int q  = lane & 15;
  const int dh = (lane >> 4) << 3;
  bf16_t* orow = O + ((size_t)b * SEQ + qrow + q) * DIM + h * HD;
#pragma unroll
  for (int t = 0; t < 4; ++t) {
#pragma unroll
    for (int g = 0; g < 2; ++g) {
      uint2 w;
      w.x = pack2(acc[t][4 * g + 0] * inv, acc[t][4 * g + 1] * inv);
      w.y = pack2(acc[t][4 * g + 2] * inv, acc[t][4 * g + 3] * inv);
      *(uint2*)(orow + 16 * t + dh + 4 * g) = w;
    }
  }
}

// ---------------------------------------------------------------------------
// Kernel 3: out = attn_out(bf16) @ proj_w + proj_b  -> fp32
// A tile is a pure bf16 copy -> async LDS DMA; B tile converts via VGPRs.
// ---------------------------------------------------------------------------
__global__ __launch_bounds__(256) void proj_gemm_kernel(
    const bf16_t* __restrict__ A, const float* __restrict__ W,
    const float* __restrict__ bias, float* __restrict__ Out) {
  __shared__ alignas(16) bf16_t As[128][40];
  __shared__ alignas(16) bf16_t Bs[128][40];
  const int mbase = blockIdx.x * 128;
  const int nbase = blockIdx.y * 128;
  const int tid   = threadIdx.x;
  const int wave  = tid >> 5;
  const int lane  = tid & 31;
  const int wm    = wave >> 2;
  const int wn    = wave & 3;

  // per-thread slots for async A-tile copy (2 x 16B per thread)
  const int ar0 = (tid >> 2);
  const int ac  = (tid & 3) << 3;
  const unsigned ldsA0 = (unsigned)(unsigned long long)&As[ar0][ac];
  const unsigned ldsA1 = (unsigned)(unsigned long long)&As[ar0 + 64][ac];

  v8f acc[4][2];
#pragma unroll
  for (int i = 0; i < 4; ++i)
#pragma unroll
    for (int j = 0; j < 2; ++j) acc[i][j] = {};

  for (int k0 = 0; k0 < DIM; k0 += 32) {
    // A tile 128x32 bf16: async global->LDS, no VGPR round trip
    {
      const unsigned long long g0 =
          (unsigned long long)(A + (size_t)(mbase + ar0) * DIM + k0 + ac);
      const unsigned long long g1 =
          (unsigned long long)(A + (size_t)(mbase + ar0 + 64) * DIM + k0 + ac);
      asm volatile("global_load_async_to_lds_b128 %0, %1, off"
                   :: "v"(ldsA0), "v"(g0) : "memory");
      asm volatile("global_load_async_to_lds_b128 %0, %1, off"
                   :: "v"(ldsA1), "v"(g1) : "memory");
    }
    // B tile: 32x128 fp32 -> bf16 transposed [n][k]; packed (k,k+1) b32 stores
#pragma unroll
    for (int p = 0; p < 2; ++p) {
      const int k = ((tid >> 5) + p * 8) << 1;
      const int n4 = (tid & 31) << 2;
      const float4 w0 = *(const float4*)(W + (size_t)(k0 + k) * DIM + nbase + n4);
      const float4 w1 = *(const float4*)(W + (size_t)(k0 + k + 1) * DIM + nbase + n4);
      *(unsigned*)&Bs[n4 + 0][k] = pack2(w0.x, w1.x);
      *(unsigned*)&Bs[n4 + 1][k] = pack2(w0.y, w1.y);
      *(unsigned*)&Bs[n4 + 2][k] = pack2(w0.z, w1.z);
      *(unsigned*)&Bs[n4 + 3][k] = pack2(w0.w, w1.w);
    }
    asm volatile("s_wait_asynccnt 0x0" ::: "memory");
    __syncthreads();
    v16bf af[4], bfr[2];
#pragma unroll
    for (int i = 0; i < 4; ++i) af[i] = load_a_rowmajor(&As[wm * 64 + 16 * i][0], 40);
#pragma unroll
    for (int j = 0; j < 2; ++j) bfr[j] = load_b_nmajor(&Bs[wn * 32 + 16 * j][0], 40);
#pragma unroll
    for (int i = 0; i < 4; ++i)
#pragma unroll
      for (int j = 0; j < 2; ++j) acc[i][j] = wmma_bf16(af[i], bfr[j], acc[i][j]);
    __syncthreads();
  }

  const int colq  = lane & 15;
  const int rhalf = (lane >> 4) << 3;
#pragma unroll
  for (int i = 0; i < 4; ++i) {
#pragma unroll
    for (int j = 0; j < 2; ++j) {
      const int n = nbase + wn * 32 + 16 * j + colq;
      const float bv = bias[n];
#pragma unroll
      for (int r8 = 0; r8 < 8; ++r8) {
        const int m = mbase + wm * 64 + 16 * i + rhalf + r8;
        Out[(size_t)m * DIM + n] = acc[i][j][r8] + bv;
      }
    }
  }
}

// ---------------------------------------------------------------------------
extern "C" void kernel_launch(void* const* d_in, const int* in_sizes, int n_in,
                              void* d_out, int out_size, void* d_ws, size_t ws_size,
                              hipStream_t stream) {
  const float* x      = (const float*)d_in[0];
  const float* qkv_w  = (const float*)d_in[1];
  const float* qkv_b  = (const float*)d_in[2];
  const float* proj_w = (const float*)d_in[3];
  const float* proj_b = (const float*)d_in[4];
  float* out = (float*)d_out;

  const size_t QKV_ELEMS = (size_t)BATCH * HEADS * SEQ * HD;  // 8,388,608
  bf16_t* qb = (bf16_t*)d_ws;                                  // 16 MB each
  bf16_t* kb = qb + QKV_ELEMS;
  bf16_t* vb = kb + QKV_ELEMS;
  bf16_t* ab = vb + QKV_ELEMS;                                 // attn out [B,SEQ,DIM]

  qkv_gemm_kernel<<<dim3(MTOT / 128, (3 * DIM) / 128), 256, 0, stream>>>(
      x, qkv_w, qkv_b, qb, kb, vb);
  flash_attn_kernel<<<dim3(BATCH * HEADS, SEQ / 128), 256, 0, stream>>>(
      qb, kb, vb, ab);
  proj_gemm_kernel<<<dim3(MTOT / 128, DIM / 128), 256, 0, stream>>>(
      ab, proj_w, proj_b, out);
}
